// ChannelAttentionModule_38989713113555
// MI455X (gfx1250) — matrix operand order repaired and checked by reading, not compiled, not run on hardware
//
#include <hip/hip_runtime.h>

// ---------------------------------------------------------------------------
// ChannelAttentionModule for MI455X (gfx1250, wave32, WMMA bf16 16x16x32)
//   B=16, H=W=64 (N=4096), C=512
//   aa  = V^T V          (per-batch 512x512x4096 GEMM)  -> f32 in ws
//   s   = softmax(aa)    (row softmax, wave32-per-row)  -> bf16 in ws
//   out = gamma * (V s) + x                              (fused epilogue)
// Register blocking: 8 waves/block, each wave owns a 32x32 f32 tile
// (4 accumulators, 2 A-frags x 2 B-frags -> 4 WMMAs per K-step).
// Second GEMM stages its A tile with global_load_async_to_lds_b128
// (ASYNCcnt) since V is K-contiguous; transposed tiles go through a
// VGPR scatter into fragment-major LDS ([row][k], pitch 40 halves).
// Workspace: v bf16 67108864 B | aa f32 16777216 B | s bf16 8388608 B
// ---------------------------------------------------------------------------

typedef __attribute__((ext_vector_type(16))) __bf16 v16bf;
typedef __attribute__((ext_vector_type(8)))  float  v8f;

#define CH    512
#define NPIX  4096
#define NB    16
#define KT    32
#define PITCH 40   // LDS row pitch in halves: even (4B pair reads) and 80B rows
                   // so row*PITCH*2 is 16B-aligned for async B128 stores

union FragU {
    unsigned int u[8];
    v16bf v;
};

__device__ __forceinline__ unsigned short f32_to_bf16(float f) {
    unsigned int b = __float_as_uint(f);
    unsigned int r = b + 0x7FFFu + ((b >> 16) & 1u);   // round to nearest even
    return (unsigned short)(r >> 16);
}

// A fragment: 16x32 bf16 (CDNA5 ISA §7.12.2):
//   lanes 0-15: M=lane, VGPR0..3 -> K pairs {0,1}..{6,7}; VGPR4..7 -> K 16..23
//   lanes 16-31: same M, K ranges shifted +8.  LDS tile stored [row][k].
__device__ __forceinline__ void load_a_frag(const unsigned short* As, int m_base,
                                            int lane, unsigned int* u) {
    const int hi = (lane >> 4) ? 8 : 0;
    const unsigned short* row = As + (m_base + (lane & 15)) * PITCH;
#pragma unroll
    for (int j = 0; j < 8; ++j) {
        int kb = (j < 4) ? (hi + 2 * j) : (16 + hi + 2 * (j - 4));
        u[j] = *reinterpret_cast<const unsigned int*>(row + kb);
    }
}

// B fragment: 32x16 bf16: lanes 0-15 -> N=lane, VGPR0..3 -> K 0..7, VGPR4..7 -> K 16..23;
// lanes 16-31 -> same N, K ranges shifted +8.  LDS tile stored [n][k].
__device__ __forceinline__ void load_b_frag(const unsigned short* Bs, int n_base,
                                            int lane, unsigned int* u) {
    const int kbase = (lane >> 4) * 8;
    const unsigned short* row = Bs + (n_base + (lane & 15)) * PITCH;
#pragma unroll
    for (int j = 0; j < 8; ++j)
        u[j] = *reinterpret_cast<const unsigned int*>(row + kbase + ((j < 4) ? 2 * j : 16 + 2 * (j - 4)));
}

__device__ __forceinline__ v8f wmma_bf16(const FragU& a, const FragU& b, v8f c) {
    return __builtin_amdgcn_wmma_f32_16x16x32_bf16(
        /*neg_a=*/false, a.v, /*neg_b=*/false, b.v,
        /*c_mod=*/(short)0, c, /*reuse_a=*/false, /*reuse_b=*/false);
}

// Async global->LDS 16B copy (CDNA5 VGLOBAL async path, tracked by ASYNCcnt).
// lds_addr: per the flat-aperture rule, addr[31:0] of a shared pointer is the
// LDS byte address.
__device__ __forceinline__ void async_copy_b128(unsigned lds_addr,
                                                const void* gaddr) {
    asm volatile("global_load_async_to_lds_b128 %0, %1, off"
                 :: "v"(lds_addr), "v"(gaddr) : "memory");
}

__device__ __forceinline__ void wait_asynccnt0() {
    asm volatile("s_wait_asynccnt 0x0" ::: "memory");
}

// ---------------------------------------------------------------------------
// Kernel 1: f32 -> bf16 conversion (8 elements / thread, 16B vector I/O)
// ---------------------------------------------------------------------------
__global__ void cam_convert_kernel(const float* __restrict__ x,
                                   unsigned short* __restrict__ v,
                                   size_t total8) {
    size_t i = (size_t)blockIdx.x * blockDim.x + threadIdx.x;
    if (i >= total8) return;
    const float4* xf = reinterpret_cast<const float4*>(x);
    float4 f0 = xf[2 * i];
    float4 f1 = xf[2 * i + 1];
    uint4 p;
    p.x = (unsigned)f32_to_bf16(f0.x) | ((unsigned)f32_to_bf16(f0.y) << 16);
    p.y = (unsigned)f32_to_bf16(f0.z) | ((unsigned)f32_to_bf16(f0.w) << 16);
    p.z = (unsigned)f32_to_bf16(f1.x) | ((unsigned)f32_to_bf16(f1.y) << 16);
    p.w = (unsigned)f32_to_bf16(f1.z) | ((unsigned)f32_to_bf16(f1.w) << 16);
    reinterpret_cast<uint4*>(v)[i] = p;
}

// ---------------------------------------------------------------------------
// Kernel 2: per-batch gram  aa[c,d] = sum_n v[n,c] v[n,d]
// Block 256 = 8 waves (4m x 2n); 128x64 tile; wave -> 32x32 subtile.
// grid = (C/64, C/128, B)
// ---------------------------------------------------------------------------
__global__ void cam_gram_kernel(const unsigned short* __restrict__ v,
                                float* __restrict__ aa) {
    __shared__ __align__(16) unsigned short As[128 * PITCH];
    __shared__ __align__(16) unsigned short Bs[64 * PITCH];

    const int bz  = blockIdx.z;
    const int cn0 = blockIdx.x * 64;
    const int cm0 = blockIdx.y * 128;
    const int t    = threadIdx.x;
    const int lane = t & 31;
    const int wave = t >> 5;
    const int wm   = (wave >> 1) * 32;
    const int wn   = (wave & 1) * 32;

    const unsigned short* vb = v + (size_t)bz * NPIX * CH;

    // staging indices: both tiles are channel-contiguous in global memory,
    // scattered to fragment-major LDS ([row][k]).
    const int sk  = t >> 3;        // 0..31 (K index)
    const int smA = (t & 7) * 16;  // 0..112 (A row group, 16 rows)
    const int smB = (t & 7) * 8;   // 0..56  (B row group, 8 rows)

    v8f acc00 = {}, acc01 = {}, acc10 = {}, acc11 = {};

    for (int k0 = 0; k0 < NPIX; k0 += KT) {
        const size_t gbase = (size_t)(k0 + sk) * CH;
        uint4 ra0 = *reinterpret_cast<const uint4*>(vb + gbase + cm0 + smA);
        uint4 ra1 = *reinterpret_cast<const uint4*>(vb + gbase + cm0 + smA + 8);
        uint4 rb  = *reinterpret_cast<const uint4*>(vb + gbase + cn0 + smB);
        if (k0 + KT < NPIX) {
            __builtin_prefetch(vb + gbase + (size_t)KT * CH + cm0 + smA, 0, 0);
            __builtin_prefetch(vb + gbase + (size_t)KT * CH + cn0 + smB, 0, 0);
        }
        __syncthreads();   // previous tile fully consumed
        unsigned int wa0[4] = {ra0.x, ra0.y, ra0.z, ra0.w};
        unsigned int wa1[4] = {ra1.x, ra1.y, ra1.z, ra1.w};
        unsigned int wbv[4] = {rb.x, rb.y, rb.z, rb.w};
#pragma unroll
        for (int i = 0; i < 4; ++i) {
            As[(smA + 2 * i)         * PITCH + sk] = (unsigned short)(wa0[i] & 0xffffu);
            As[(smA + 2 * i + 1)     * PITCH + sk] = (unsigned short)(wa0[i] >> 16);
            As[(smA + 8 + 2 * i)     * PITCH + sk] = (unsigned short)(wa1[i] & 0xffffu);
            As[(smA + 8 + 2 * i + 1) * PITCH + sk] = (unsigned short)(wa1[i] >> 16);
            Bs[(smB + 2 * i)         * PITCH + sk] = (unsigned short)(wbv[i] & 0xffffu);
            Bs[(smB + 2 * i + 1)     * PITCH + sk] = (unsigned short)(wbv[i] >> 16);
        }
        __syncthreads();

        FragU fa0, fa1, fb0, fb1;
        load_a_frag(As, wm,      lane, fa0.u);
        load_a_frag(As, wm + 16, lane, fa1.u);
        load_b_frag(Bs, wn,      lane, fb0.u);
        load_b_frag(Bs, wn + 16, lane, fb1.u);
        acc00 = wmma_bf16(fa0, fb0, acc00);
        acc01 = wmma_bf16(fa0, fb1, acc01);
        acc10 = wmma_bf16(fa1, fb0, acc10);
        acc11 = wmma_bf16(fa1, fb1, acc11);
    }

    // C/D layout: VGPR r -> row = (lane>>4)*8 + r, col = lane&15
    const int r0  = cm0 + wm + (lane >> 4) * 8;
    const int col = cn0 + wn + (lane & 15);
#pragma unroll
    for (int r = 0; r < 8; ++r) {
        size_t i0 = ((size_t)bz * CH + r0 + r) * CH + col;
        size_t i1 = ((size_t)bz * CH + r0 + 16 + r) * CH + col;
        aa[i0]      = acc00[r];
        aa[i0 + 16] = acc01[r];
        aa[i1]      = acc10[r];
        aa[i1 + 16] = acc11[r];
    }
}

// ---------------------------------------------------------------------------
// Kernel 3: row softmax over aa, write bf16 s.  One wave32 per 512-row.
// grid = B*C/8, block = 256 (8 waves)
// ---------------------------------------------------------------------------
__global__ void cam_softmax_kernel(const float* __restrict__ aa,
                                   unsigned short* __restrict__ s) {
    const int lane = threadIdx.x & 31;
    const int wave = threadIdx.x >> 5;
    const size_t row = (size_t)blockIdx.x * 8 + wave;   // < NB*CH
    const float* a = aa + row * CH;

    float vals[16];
    float vmax = -3.0e38f;
#pragma unroll
    for (int i = 0; i < 16; ++i) {
        vals[i] = a[lane + 32 * i];
        vmax = fmaxf(vmax, vals[i]);
    }
#pragma unroll
    for (int off = 16; off > 0; off >>= 1)
        vmax = fmaxf(vmax, __shfl_xor(vmax, off, 32));

    float sum = 0.f;
#pragma unroll
    for (int i = 0; i < 16; ++i) {
        vals[i] = __expf(vals[i] - vmax);
        sum += vals[i];
    }
#pragma unroll
    for (int off = 16; off > 0; off >>= 1)
        sum += __shfl_xor(sum, off, 32);

    const float inv = 1.0f / sum;
    unsigned short* so = s + row * CH;
#pragma unroll
    for (int i = 0; i < 16; ++i)
        so[lane + 32 * i] = f32_to_bf16(vals[i] * inv);
}

// ---------------------------------------------------------------------------
// Kernel 4: aaa = V s, out = gamma*aaa + x (fused).
// Block 256 = 8 waves (4m x 2n); 128(M) x 64(N) tile; K = C = 512.
// A tile (K-contiguous) staged via global_load_async_to_lds_b128.
// grid = (NPIX/128, C/64, B)
// ---------------------------------------------------------------------------
__global__ void cam_apply_kernel(const unsigned short* __restrict__ v,
                                 const unsigned short* __restrict__ s,
                                 const float* __restrict__ x,
                                 const float* __restrict__ gamma,
                                 float* __restrict__ out) {
    __shared__ __align__(16) unsigned short As[128 * PITCH];
    __shared__ __align__(16) unsigned short Bs[64 * PITCH];

    const int bz  = blockIdx.z;
    const int m0g = blockIdx.x * 128;  // pixel rows
    const int n0g = blockIdx.y * 64;   // output channels
    const int t    = threadIdx.x;
    const int lane = t & 31;
    const int wave = t >> 5;
    const int wm   = (wave >> 1) * 32;
    const int wn   = (wave & 1) * 32;

    const unsigned short* vb = v + (size_t)bz * NPIX * CH;
    const unsigned short* sb = s + (size_t)bz * CH * CH;

    // A staging (async): thread -> one row m, 16-half chunk (two B128 copies)
    const int am = t >> 1;         // 0..127
    const int ak = (t & 1) * 16;   // 0 or 16
    const unsigned lds_a = (unsigned)(size_t)(As + am * PITCH + ak);
    // B staging: 8 contiguous n at one k; transpose scatter to Bs[n][k]
    const int bk = t >> 3;         // 0..31
    const int bn = (t & 7) * 8;    // 0..56

    v8f acc00 = {}, acc01 = {}, acc10 = {}, acc11 = {};

    for (int k0 = 0; k0 < CH; k0 += KT) {
        uint4 rb = *reinterpret_cast<const uint4*>(sb + (size_t)(k0 + bk) * CH + n0g + bn);
        __syncthreads();   // previous tile fully consumed
        // A tile: direct global->LDS async copies (32B per thread)
        const unsigned short* ga = vb + (size_t)(m0g + am) * CH + k0 + ak;
        async_copy_b128(lds_a,      ga);
        async_copy_b128(lds_a + 16, ga + 8);
        // B tile: transpose scatter
        unsigned int wbv[4] = {rb.x, rb.y, rb.z, rb.w};
#pragma unroll
        for (int i = 0; i < 4; ++i) {
            Bs[(bn + 2 * i)     * PITCH + bk] = (unsigned short)(wbv[i] & 0xffffu);
            Bs[(bn + 2 * i + 1) * PITCH + bk] = (unsigned short)(wbv[i] >> 16);
        }
        wait_asynccnt0();
        __syncthreads();

        FragU fa0, fa1, fb0, fb1;
        load_a_frag(As, wm,      lane, fa0.u);
        load_a_frag(As, wm + 16, lane, fa1.u);
        load_b_frag(Bs, wn,      lane, fb0.u);
        load_b_frag(Bs, wn + 16, lane, fb1.u);
        acc00 = wmma_bf16(fa0, fb0, acc00);
        acc01 = wmma_bf16(fa0, fb1, acc01);
        acc10 = wmma_bf16(fa1, fb0, acc10);
        acc11 = wmma_bf16(fa1, fb1, acc11);
    }

    const float g = gamma[0];
    const int r0  = m0g + wm + (lane >> 4) * 8;
    const int col = n0g + wn + (lane & 15);
#pragma unroll
    for (int r = 0; r < 8; ++r) {
        size_t i0 = ((size_t)bz * NPIX + r0 + r) * CH + col;
        size_t i1 = ((size_t)bz * NPIX + r0 + 16 + r) * CH + col;
        out[i0]      = g * acc00[r] + x[i0];
        out[i0 + 16] = g * acc01[r] + x[i0 + 16];
        out[i1]      = g * acc10[r] + x[i1];
        out[i1 + 16] = g * acc11[r] + x[i1 + 16];
    }
}

// ---------------------------------------------------------------------------
extern "C" void kernel_launch(void* const* d_in, const int* in_sizes, int n_in,
                              void* d_out, int out_size, void* d_ws, size_t ws_size,
                              hipStream_t stream) {
    const float* x     = (const float*)d_in[0];
    const float* gamma = (const float*)d_in[1];
    float* out = (float*)d_out;

    char* ws = (char*)d_ws;
    unsigned short* v  = (unsigned short*)ws;                                  // 67108864 B
    float*          aa = (float*)(ws + (size_t)67108864);                      // 16777216 B
    unsigned short* s  = (unsigned short*)(ws + (size_t)67108864 + 16777216);  // 8388608 B

    const size_t total  = (size_t)NB * NPIX * CH;   // 33.55M elements
    const size_t total8 = total / 8;

    cam_convert_kernel<<<dim3((unsigned)((total8 + 255) / 256)), 256, 0, stream>>>(x, v, total8);
    cam_gram_kernel<<<dim3(CH / 64, CH / 128, NB), 256, 0, stream>>>(v, aa);
    cam_softmax_kernel<<<dim3(NB * CH / 8), 256, 0, stream>>>(aa, s);
    cam_apply_kernel<<<dim3(NPIX / 128, CH / 64, NB), 256, 0, stream>>>(v, s, x, gamma, out);
}
